// GCN_18820546691816
// MI455X (gfx1250) — matrix operand order repaired and checked by reading, not compiled, not run on hardware
//
#include <hip/hip_runtime.h>
#include <hip/hip_bf16.h>

typedef __attribute__((ext_vector_type(16))) __bf16 v16bf;
typedef __attribute__((ext_vector_type(8)))  float  v8f;

#define WAVES 8
#define TPB   (WAVES * 32)

// Wave-level phase fence for the per-wave LDS stage buffer: orders this wave's
// LDS stores vs. subsequent LDS loads (cross-lane exchange) and stops the
// compiler from hoisting stage loads across stage stores.
#define WAVE_LDS_FENCE() asm volatile("s_wait_dscnt 0" ::: "memory")

union ABFrag { v16bf v; unsigned int u[8]; };

__device__ __forceinline__ unsigned short f2bf(float f) {
  unsigned int u = __float_as_uint(f);
  u += 0x7FFFu + ((u >> 16) & 1u);          // round-to-nearest-even
  return (unsigned short)(u >> 16);
}
__device__ __forceinline__ unsigned int pack2(float lo, float hi) {
  return (unsigned int)f2bf(lo) | ((unsigned int)f2bf(hi) << 16);
}
// A-matrix K offset per VGPR v (ISA 16-bit A layout): v<4 -> 2v, v>=4 -> 16+2(v-4)
__device__ __forceinline__ int koffA(int v) { return (v < 4) ? 2 * v : 8 + 2 * v; }

#define WMMA_BF16(A, B, C) \
  __builtin_amdgcn_wmma_f32_16x16x32_bf16(false, (A), false, (B), (short)0, (C), false, false)

// ---- fragment builders -----------------------------------------------------

// B fragment from LDS weights stored as [Npad][Kpad] bf16 (K contiguous).
// B layout: lane holds column n=lane%16; VGPR v holds K=koffA(v),+1 (+8 for lanes>=16).
__device__ __forceinline__ v16bf loadB(const unsigned short* W, int Kpad, int n, int kbase) {
  const unsigned int* w32 = (const unsigned int*)W;
  ABFrag t;
  int d = (n * Kpad + (kbase & ~31) + ((kbase >> 4) & 1) * 8) >> 1;
#pragma unroll
  for (int v = 0; v < 8; ++v) t.u[v] = w32[d + (koffA(v) >> 1)];
  return t.v;
}

// A fragment from per-wave LDS stage buffer [16][64] bf16.
__device__ __forceinline__ v16bf loadAstage(const unsigned short* stg, int m, int c, int half) {
  const unsigned int* s32 = (const unsigned int*)stg;
  ABFrag t;
  int base = m * 64 + c * 32 + 8 * half;
#pragma unroll
  for (int v = 0; v < 8; ++v) t.u[v] = s32[(base + koffA(v)) >> 1];
  return t.v;
}

// A fragment straight from global f32 input row (K padded 112 -> 128 with zeros).
__device__ __forceinline__ v16bf loadAx(const float* xrow, int c, int half) {
  ABFrag t;
#pragma unroll
  for (int v = 0; v < 8; ++v) {
    int K = c * 32 + koffA(v) + 8 * half;
    if (K < 112) {
      const float2 f = *(const float2*)(xrow + K);
      t.u[v] = pack2(f.x, f.y);
    } else {
      t.u[v] = 0u;
    }
  }
  return t.v;
}

// D fragment (16x16 f32: VGPR r -> row r + 8*half, col lane%16) -> bias + ReLU -> stage bf16
__device__ __forceinline__ void storeStageV(unsigned short* stg, const v8f& acc, float bv,
                                            int t, int l16, int half) {
  const int col = t * 16 + l16;
#pragma unroll
  for (int r = 0; r < 8; ++r) {
    float val = acc[r] + bv;
    val = val > 0.0f ? val : 0.0f;
    stg[(r + 8 * half) * 64 + col] = f2bf(val);
  }
}

// ---- setup: fused bf16 weights + padded biases into workspace --------------

__global__ void gcn_setup(const int* __restrict__ edge,
                          const float* __restrict__ W1, const float* __restrict__ b1,
                          const float* __restrict__ W2, const float* __restrict__ b2,
                          const float* __restrict__ Wl1, const float* __restrict__ bl1,
                          const float* __restrict__ Wl2, const float* __restrict__ bl2,
                          unsigned short* __restrict__ wsu, float* __restrict__ wsb) {
  __shared__ float sA[49];
  if (threadIdx.x == 0) {
    float deg[7], dinv[7];
    for (int n = 0; n < 7; ++n) deg[n] = 1.0f;            // self loop
    for (int e = 0; e < 14; ++e) deg[edge[14 + e] % 7] += 1.0f;
    for (int n = 0; n < 7; ++n) dinv[n] = rsqrtf(deg[n]);
    for (int i = 0; i < 49; ++i) sA[i] = 0.0f;
    for (int e = 0; e < 14; ++e) {
      int s = edge[e] % 7, d = edge[14 + e] % 7;
      sA[d * 7 + s] += dinv[s] * dinv[d];
    }
    for (int n = 0; n < 7; ++n) sA[n * 7 + n] += dinv[n] * dinv[n];
  }
  __syncthreads();

  // M1 = kron(A, W1^T): [128K x 64N] pad, stored [N][K] bf16
  for (int i = threadIdx.x; i < 64 * 128; i += blockDim.x) {
    int n = i >> 7, k = i & 127;
    float v = 0.0f;
    if (n < 56 && k < 112) v = sA[(n >> 3) * 7 + (k >> 4)] * W1[(k & 15) * 8 + (n & 7)];
    wsu[i] = f2bf(v);
  }
  // M2 = kron(A, W2^T): [64K x 64N] pad
  for (int i = threadIdx.x; i < 64 * 64; i += blockDim.x) {
    int n = i >> 6, k = i & 63;
    float v = 0.0f;
    if (n < 56 && k < 56) v = sA[(n >> 3) * 7 + (k >> 3)] * W2[(k & 7) * 8 + (n & 7)];
    wsu[8192 + i] = f2bf(v);
  }
  // Wl1: [64K x 32N] pad
  for (int i = threadIdx.x; i < 32 * 64; i += blockDim.x) {
    int n = i >> 6, k = i & 63;
    float v = 0.0f;
    if (n < 24 && k < 56) v = Wl1[k * 24 + n];
    wsu[12288 + i] = f2bf(v);
  }
  // Wl2: [32K x 16N] pad (only column 0 live)
  for (int i = threadIdx.x; i < 16 * 32; i += blockDim.x) {
    int n = i >> 5, k = i & 31;
    float v = 0.0f;
    if (n == 0 && k < 24) v = Wl2[k];
    wsu[14336 + i] = f2bf(v);
  }
  // padded biases
  for (int i = threadIdx.x; i < 64; i += blockDim.x) {
    wsb[i]      = (i < 56) ? b1[i & 7] : 0.0f;
    wsb[64 + i] = (i < 56) ? b2[i & 7] : 0.0f;
  }
  for (int i = threadIdx.x; i < 32; i += blockDim.x) wsb[128 + i] = (i < 24) ? bl1[i] : 0.0f;
  if (threadIdx.x == 0) wsb[160] = bl2[0];
}

// ---- main: streaming bf16-WMMA MLP over 16-graph tiles ---------------------

__global__ __launch_bounds__(TPB) void gcn_main(const float* __restrict__ x1,
                                                float* __restrict__ out,
                                                const unsigned short* __restrict__ wsu,
                                                const float* __restrict__ wsb, int nTiles) {
  __shared__ unsigned short sW1[64 * 128];
  __shared__ unsigned short sW2[64 * 64];
  __shared__ unsigned short sW3[32 * 64];
  __shared__ unsigned short sW4[16 * 32];
  __shared__ float sBias[161];
  __shared__ unsigned short sStage[WAVES * 16 * 64];

  for (int i = threadIdx.x; i < 64 * 128; i += TPB) sW1[i] = wsu[i];
  for (int i = threadIdx.x; i < 64 * 64; i += TPB) sW2[i] = wsu[8192 + i];
  for (int i = threadIdx.x; i < 32 * 64; i += TPB) sW3[i] = wsu[12288 + i];
  for (int i = threadIdx.x; i < 16 * 32; i += TPB) sW4[i] = wsu[14336 + i];
  for (int i = threadIdx.x; i < 161; i += TPB) sBias[i] = wsb[i];
  __syncthreads();

  const int lane = threadIdx.x & 31;
  const int wv   = threadIdx.x >> 5;
  const int half = lane >> 4;
  const int l16  = lane & 15;
  unsigned short* stg = &sStage[wv * 1024];

  // Hoist loop-invariant B fragments for layers 2..4 into registers (13 frags).
  v16bf B2[2][4], B3[2][2], B4;
#pragma unroll
  for (int c = 0; c < 2; ++c)
#pragma unroll
    for (int t = 0; t < 4; ++t) B2[c][t] = loadB(sW2, 64, t * 16 + l16, c * 32 + 16 * half);
#pragma unroll
  for (int c = 0; c < 2; ++c)
#pragma unroll
    for (int t = 0; t < 2; ++t) B3[c][t] = loadB(sW3, 64, t * 16 + l16, c * 32 + 16 * half);
  B4 = loadB(sW4, 32, l16, 16 * half);

  // Hoist per-lane bias scalars.
  float bias1v[4], bias2v[4], bias3v[2];
#pragma unroll
  for (int t = 0; t < 4; ++t) {
    bias1v[t] = sBias[t * 16 + l16];
    bias2v[t] = sBias[64 + t * 16 + l16];
  }
#pragma unroll
  for (int t = 0; t < 2; ++t) bias3v[t] = sBias[128 + t * 16 + l16];
  const float bl2v = sBias[160];

  for (int tile = blockIdx.x * WAVES + wv; tile < nTiles; tile += (int)gridDim.x * WAVES) {
    const float* xrow = x1 + (size_t)tile * (16 * 112) + (size_t)l16 * 112;

    // Layer 1: [16x128] x [128x64]  (B fragments gathered from LDS weights)
    v8f acc[4] = {};
#pragma unroll
    for (int c = 0; c < 4; ++c) {
      v16bf a = loadAx(xrow, c, half);
#pragma unroll
      for (int t = 0; t < 4; ++t) {
        v16bf b = loadB(sW1, 128, t * 16 + l16, c * 32 + 16 * half);
        acc[t] = WMMA_BF16(a, b, acc[t]);
      }
    }
    WAVE_LDS_FENCE();   // prev-iteration stage reads done before overwrite
#pragma unroll
    for (int t = 0; t < 4; ++t) storeStageV(stg, acc[t], bias1v[t], t, l16, half);
    WAVE_LDS_FENCE();   // stage writes visible before reads

    // Layer 2: [16x64] x [64x64]
    v8f acc2[4] = {};
#pragma unroll
    for (int c = 0; c < 2; ++c) {
      v16bf a = loadAstage(stg, l16, c, half);
#pragma unroll
      for (int t = 0; t < 4; ++t) acc2[t] = WMMA_BF16(a, B2[c][t], acc2[t]);
    }
    WAVE_LDS_FENCE();
#pragma unroll
    for (int t = 0; t < 4; ++t) storeStageV(stg, acc2[t], bias2v[t], t, l16, half);
    WAVE_LDS_FENCE();

    // Layer 3: [16x64] x [64x32]
    v8f acc3[2] = {};
#pragma unroll
    for (int c = 0; c < 2; ++c) {
      v16bf a = loadAstage(stg, l16, c, half);
#pragma unroll
      for (int t = 0; t < 2; ++t) acc3[t] = WMMA_BF16(a, B3[c][t], acc3[t]);
    }
    WAVE_LDS_FENCE();
#pragma unroll
    for (int t = 0; t < 2; ++t) storeStageV(stg, acc3[t], bias3v[t], t, l16, half);
    WAVE_LDS_FENCE();

    // Layer 4: [16x32] x [32x16] (only N=0 valid)
    v8f acc4 = {};
    {
      v16bf a = loadAstage(stg, l16, 0, half);
      acc4 = WMMA_BF16(a, B4, acc4);
    }
    if (l16 == 0) {
      float* o = out + (size_t)tile * 16 + 8 * half;
#pragma unroll
      for (int r = 0; r < 8; ++r) o[r] = acc4[r] + bl2v;
    }
  }
}

extern "C" void kernel_launch(void* const* d_in, const int* in_sizes, int n_in,
                              void* d_out, int out_size, void* d_ws, size_t ws_size,
                              hipStream_t stream) {
  const float* x1  = (const float*)d_in[0];
  const int*   edg = (const int*)d_in[1];
  const float* W1  = (const float*)d_in[2];
  const float* b1  = (const float*)d_in[3];
  const float* W2  = (const float*)d_in[4];
  const float* b2  = (const float*)d_in[5];
  const float* Wl1 = (const float*)d_in[6];
  const float* bl1 = (const float*)d_in[7];
  const float* Wl2 = (const float*)d_in[8];
  const float* bl2 = (const float*)d_in[9];

  unsigned short* wsu = (unsigned short*)d_ws;
  float* wsb = (float*)((char*)d_ws + 14848 * sizeof(unsigned short));

  const int B = in_sizes[0] / 112;
  const int nTiles = B / 16;

  hipLaunchKernelGGL(gcn_setup, dim3(1), dim3(256), 0, stream,
                     edg, W1, b1, W2, b2, Wl1, bl1, Wl2, bl2, wsu, wsb);

  int blocks = (nTiles + WAVES - 1) / WAVES;
  if (blocks > 512) blocks = 512;
  if (blocks < 1) blocks = 1;
  hipLaunchKernelGGL(gcn_main, dim3(blocks), dim3(TPB), 0, stream,
                     x1, (float*)d_out, wsu, wsb, nTiles);
}
